// Attention_31542239822413
// MI455X (gfx1250) — matrix operand order repaired and checked by reading, not compiled, not run on hardware
//
#include <hip/hip_runtime.h>
#include <hip/hip_bf16.h>

#define DIMC   768
#define NPATCH 1024
#define NH     12
#define DH     64
#define BB     8

typedef __attribute__((ext_vector_type(16))) __bf16 bf16x16;
typedef __attribute__((ext_vector_type(8)))  float  f32x8;
typedef __attribute__((ext_vector_type(4)))  int    v4i;

// gfx1250 async global->LDS copy path (ASYNCcnt-tracked), guarded so the file
// compiles on toolchains that don't declare the builtins.
#if defined(__AMDGCN__) && __has_builtin(__builtin_amdgcn_global_load_async_to_lds_b128)
#define HAVE_ASYNC_LDS 1
#else
#define HAVE_ASYNC_LDS 0
#endif

#if HAVE_ASYNC_LDS
// Builtin signature wants v4i* in AS1 (global) and AS3 (LDS).
#define GLOBAL_AS(p) ((__attribute__((address_space(1))) v4i*)(p))
#define LDS_AS(p)    ((__attribute__((address_space(3))) v4i*)(p))
__device__ __forceinline__ void wait_async0() {
#if __has_builtin(__builtin_amdgcn_s_wait_asynccnt)
  __builtin_amdgcn_s_wait_asynccnt(0);
#else
  asm volatile("s_wait_asynccnt 0" ::: "memory");
#endif
}
#endif

union FragU { bf16x16 v; unsigned u[8]; };

__device__ __forceinline__ f32x8 zero8() {
  f32x8 z = {0.f, 0.f, 0.f, 0.f, 0.f, 0.f, 0.f, 0.f};
  return z;
}

__device__ __forceinline__ f32x8 wmma_bf16(bf16x16 a, bf16x16 b, f32x8 c) {
  // D = A(16x32 bf16) * B(32x16 bf16) + C(16x16 f32)
  return __builtin_amdgcn_wmma_f32_16x16x32_bf16(false, a, false, b, (short)0, c,
                                                 false, false);
}

// A-matrix fragment per ISA 7.12.2 (16-bit A 16x32): lane = (hf,row), VGPR v
// holds k-pair; v<4 -> k = 2v + 8*hf, v>=4 -> k = 16 + 2(v-4) + 8*hf.
// LDS tile is row-major [row][k], element stride `stride` (even).
__device__ __forceinline__ bf16x16 load_a_frag(const __bf16* lds, int row,
                                               int stride, int hf) {
  FragU f;
  const __bf16* base = lds + row * stride + 8 * hf;
  #pragma unroll
  for (int v = 0; v < 8; ++v) {
    int kb = (v < 4) ? (2 * v) : (8 + 2 * v);   // 2v  or  16 + 2(v-4)
    f.u[v] = *(const unsigned*)(base + kb);
  }
  return f.v;
}

// B-matrix fragment (16-bit B 32x16): lane = (hf,col), same k order as A frag.
// LDS tile stored transposed as Bt[col][k] so k-pairs are dword-contiguous.
__device__ __forceinline__ bf16x16 load_bt_frag(const __bf16* lds, int col,
                                                int stride, int hf) {
  FragU f;
  const __bf16* base = lds + col * stride + 8 * hf;
  #pragma unroll
  for (int v = 0; v < 8; ++v)
    f.u[v] = *(const unsigned*)(base + ((v < 4) ? (2 * v) : (8 + 2 * v)));
  return f.v;
}

// ---------------------------------------------------------------------------
// Kernel 1: qkv = x @ W_qkv   (f32 in -> bf16 WMMA -> bf16 out, head layout)
// Grid: (8192/64, 2304/64); 256 threads = 8 waves; each wave: 16 rows x 64 cols
// ---------------------------------------------------------------------------
__global__ __launch_bounds__(256) void qkv_gemm_kernel(
    const float* __restrict__ x, const float* __restrict__ Wqkv,
    __bf16* __restrict__ qkv) {
  __shared__ __align__(16) __bf16 As[64 * 40];
  __shared__ __align__(16) __bf16 Bs[64 * 40];
  const int t = threadIdx.x, wv = t >> 5, lane = t & 31;
  const int hf = lane >> 4, lc = lane & 15;
  const int m0 = blockIdx.x * 64;
  const int nblk = blockIdx.y;
  const int n0 = nblk * 64;
  f32x8 acc[4] = {zero8(), zero8(), zero8(), zero8()};

  for (int kk = 0; kk < DIMC; kk += 32) {
    __syncthreads();
    #pragma unroll
    for (int i = 0; i < 8; ++i) {              // A tile 64x32, row-major
      int idx = t + i * 256;
      int r = idx >> 5, c = idx & 31;
      As[r * 40 + c] = (__bf16)x[(size_t)(m0 + r) * DIMC + kk + c];
    }
    #pragma unroll
    for (int i = 0; i < 8; ++i) {              // B tile 32x64 -> Bt[col][k]
      int idx = t + i * 256;
      int k = idx >> 6, c = idx & 63;
      Bs[c * 40 + k] = (__bf16)Wqkv[(size_t)(kk + k) * (3 * DIMC) + n0 + c];
    }
    __syncthreads();
    bf16x16 a = load_a_frag(As, wv * 16 + lc, 40, hf);
    #pragma unroll
    for (int tt = 0; tt < 4; ++tt)
      acc[tt] = wmma_bf16(a, load_bt_frag(Bs, tt * 16 + lc, 40, hf), acc[tt]);
  }

  // 64-col block == exactly one (q|k|v, head) pair since 768 = 12*64.
  const int which = nblk / NH;
  const int h = nblk % NH;
  __bf16* dst = qkv + (size_t)which * ((size_t)BB * NH * NPATCH * DH);
  #pragma unroll
  for (int tt = 0; tt < 4; ++tt)
    #pragma unroll
    for (int i = 0; i < 8; ++i) {
      int m = m0 + wv * 16 + i + 8 * hf;       // D layout: row = i + 8*hf
      int b = m >> 10, n = m & (NPATCH - 1);
      dst[(((size_t)(b * NH + h)) * NPATCH + n) * DH + tt * 16 + lc] =
          (__bf16)acc[tt][i];
    }
}

// ---------------------------------------------------------------------------
// Kernel 2: flash attention, one WG (4 waves) per (b, h, 64 query rows).
// rel_index[n,m] = m - n + 1023 computed analytically (skip 4MB gather input).
// Q/K tiles use async global->LDS b128 copies when the toolchain exposes them.
// ---------------------------------------------------------------------------
__global__ __launch_bounds__(128) void flash_attn_kernel(
    const __bf16* __restrict__ qkv, const float* __restrict__ bias_table,
    __bf16* __restrict__ o_ws) {
  __shared__ __align__(16) __bf16 Qs[64 * 72];
  __shared__ __align__(16) __bf16 Ks[64 * 72];        // [key][d]  == Bt for QK^T
  __shared__ __align__(16) __bf16 Vt[64 * 72];        // [d][key]  == Bt for P*V
  __shared__ __align__(16) __bf16 Ps[4 * 16 * 72];    // per-wave P tile 16x64
  const int t = threadIdx.x, wv = t >> 5, lane = t & 31;
  const int hf = lane >> 4, lc = lane & 15;
  const int qblk = blockIdx.x & 15;
  const int bh = blockIdx.x >> 4;                     // 0..95
  const int b = bh / NH, h = bh % NH;
  const int q0 = qblk * 64;
  const size_t plane = (size_t)BB * NH * NPATCH * DH;
  const __bf16* qg = qkv + (size_t)bh * NPATCH * DH;
  const __bf16* kg = qg + plane;
  const __bf16* vg = qg + 2 * plane;

#if HAVE_ASYNC_LDS
  #pragma unroll
  for (int i = 0; i < 4; ++i) {                       // Q block: 512 b128 chunks
    int c = t + i * 128;
    int r = c >> 3, c8 = (c & 7) * 8;
    __builtin_amdgcn_global_load_async_to_lds_b128(
        GLOBAL_AS(qg + (size_t)(q0 + r) * DH + c8), LDS_AS(Qs + r * 72 + c8),
        0, 0);
  }
  wait_async0();
#else
  #pragma unroll
  for (int i = 0; i < 32; ++i) {                      // Q block 64x64
    int idx = t + i * 128;
    int r = idx >> 6, d = idx & 63;
    Qs[r * 72 + d] = qg[(size_t)(q0 + r) * DH + d];
  }
#endif
  __syncthreads();
  bf16x16 qf0 = load_a_frag(Qs, wv * 16 + lc, 72, hf);        // k = d 0..31
  bf16x16 qf1 = load_a_frag(Qs + 32, wv * 16 + lc, 72, hf);   // k = d 32..63

  f32x8 accO[4] = {zero8(), zero8(), zero8(), zero8()};
  float mrow[8], lrow[8];
  #pragma unroll
  for (int i = 0; i < 8; ++i) { mrow[i] = -1e30f; lrow[i] = 0.f; }
  __bf16* pw = Ps + wv * 16 * 72;

  for (int kb = 0; kb < NPATCH; kb += 64) {
    __syncthreads();                                  // prev iter done w/ Ks,Vt
#if HAVE_ASYNC_LDS
    #pragma unroll
    for (int i = 0; i < 4; ++i) {                     // K tile via ASYNCcnt DMA
      int c = t + i * 128;
      int r = c >> 3, c8 = (c & 7) * 8;
      __builtin_amdgcn_global_load_async_to_lds_b128(
          GLOBAL_AS(kg + (size_t)(kb + r) * DH + c8), LDS_AS(Ks + r * 72 + c8),
          0, 0);
    }
    #pragma unroll
    for (int i = 0; i < 32; ++i) {                    // V transpose (manual)
      int idx = t + i * 128;
      int r = idx >> 6, d = idx & 63;
      Vt[d * 72 + r] = vg[(size_t)(kb + r) * DH + d];
    }
    wait_async0();
#else
    #pragma unroll
    for (int i = 0; i < 32; ++i) {
      int idx = t + i * 128;
      int r = idx >> 6, d = idx & 63;
      Ks[r * 72 + d] = kg[(size_t)(kb + r) * DH + d];
      Vt[d * 72 + r] = vg[(size_t)(kb + r) * DH + d];
    }
#endif
    __syncthreads();

    f32x8 s[4];
    #pragma unroll
    for (int tt = 0; tt < 4; ++tt) {                  // S = Q K^T (16 x 64)
      f32x8 z = zero8();
      z = wmma_bf16(qf0, load_bt_frag(Ks, tt * 16 + lc, 72, hf), z);
      z = wmma_bf16(qf1, load_bt_frag(Ks + 32, tt * 16 + lc, 72, hf), z);
      s[tt] = z;
    }
    #pragma unroll
    for (int tt = 0; tt < 4; ++tt) {                  // scale + rel-pos bias
      int key = kb + tt * 16 + lc;
      #pragma unroll
      for (int i = 0; i < 8; ++i) {
        int n = q0 + wv * 16 + i + 8 * hf;
        s[tt][i] = s[tt][i] * 0.125f +
                   bias_table[(size_t)(key - n + (NPATCH - 1)) * NH + h];
      }
    }
    float mc[8];                                      // row max (16 lanes/row)
    #pragma unroll
    for (int i = 0; i < 8; ++i)
      mc[i] = fmaxf(fmaxf(s[0][i], s[1][i]), fmaxf(s[2][i], s[3][i]));
    #pragma unroll
    for (int mk = 1; mk <= 8; mk <<= 1)
      #pragma unroll
      for (int i = 0; i < 8; ++i)
        mc[i] = fmaxf(mc[i], __shfl_xor(mc[i], mk, 32));
    float alpha[8], psum[8];
    #pragma unroll
    for (int i = 0; i < 8; ++i) {
      float mn = fmaxf(mrow[i], mc[i]);
      alpha[i] = exp2f((mrow[i] - mn) * 1.44269504f);
      mrow[i] = mn;
      psum[i] = 0.f;
    }
    #pragma unroll
    for (int tt = 0; tt < 4; ++tt)
      #pragma unroll
      for (int i = 0; i < 8; ++i) {
        float p = exp2f((s[tt][i] - mrow[i]) * 1.44269504f);
        psum[i] += p;
        pw[(i + 8 * hf) * 72 + tt * 16 + lc] = (__bf16)p;   // C-layout -> LDS
      }
    #pragma unroll
    for (int i = 0; i < 8; ++i) lrow[i] = alpha[i] * lrow[i] + psum[i];
    #pragma unroll
    for (int tt = 0; tt < 4; ++tt)
      #pragma unroll
      for (int i = 0; i < 8; ++i) accO[tt][i] *= alpha[i];
    __syncthreads();                                  // P visible to all lanes
    bf16x16 pa0 = load_a_frag(pw, lc, 72, hf);        // P as A-operand
    bf16x16 pa1 = load_a_frag(pw + 32, lc, 72, hf);
    #pragma unroll
    for (int tt = 0; tt < 4; ++tt) {                  // O += P * V
      accO[tt] = wmma_bf16(pa0, load_bt_frag(Vt, tt * 16 + lc, 72, hf), accO[tt]);
      accO[tt] = wmma_bf16(pa1, load_bt_frag(Vt + 32, tt * 16 + lc, 72, hf), accO[tt]);
    }
    if (kb + 64 < NPATCH)
      __builtin_prefetch(kg + (size_t)(kb + 64) * DH, 0, 0);
  }

  float linv[8];
  #pragma unroll
  for (int i = 0; i < 8; ++i) {                       // row sum of l partials
    float sum = lrow[i];
    #pragma unroll
    for (int mk = 1; mk <= 8; mk <<= 1) sum += __shfl_xor(sum, mk, 32);
    linv[i] = 1.f / sum;
  }
  #pragma unroll
  for (int tt = 0; tt < 4; ++tt)
    #pragma unroll
    for (int i = 0; i < 8; ++i) {
      int n = q0 + wv * 16 + i + 8 * hf;
      o_ws[((size_t)(b * NPATCH + n)) * DIMC + h * DH + tt * 16 + lc] =
          (__bf16)(accO[tt][i] * linv[i]);
    }
}

// ---------------------------------------------------------------------------
// Kernel 3: out = O @ W_proj + b_proj   (bf16 A already, f32 output)
// ---------------------------------------------------------------------------
__global__ __launch_bounds__(256) void proj_gemm_kernel(
    const __bf16* __restrict__ O, const float* __restrict__ Wproj,
    const float* __restrict__ bproj, float* __restrict__ out) {
  __shared__ __align__(16) __bf16 As[64 * 40];
  __shared__ __align__(16) __bf16 Bs[64 * 40];
  const int t = threadIdx.x, wv = t >> 5, lane = t & 31;
  const int hf = lane >> 4, lc = lane & 15;
  const int m0 = blockIdx.x * 64;
  const int n0 = blockIdx.y * 64;
  f32x8 acc[4] = {zero8(), zero8(), zero8(), zero8()};

  for (int kk = 0; kk < DIMC; kk += 32) {
    __syncthreads();
#if HAVE_ASYNC_LDS
    {                                           // A tile: 256 b128 chunks
      int r = t >> 2, c8 = (t & 3) * 8;
      __builtin_amdgcn_global_load_async_to_lds_b128(
          GLOBAL_AS(O + (size_t)(m0 + r) * DIMC + kk + c8),
          LDS_AS(As + r * 40 + c8), 0, 0);
    }
#else
    #pragma unroll
    for (int i = 0; i < 8; ++i) {
      int idx = t + i * 256;
      int r = idx >> 5, c = idx & 31;
      As[r * 40 + c] = O[(size_t)(m0 + r) * DIMC + kk + c];
    }
#endif
    #pragma unroll
    for (int i = 0; i < 8; ++i) {
      int idx = t + i * 256;
      int k = idx >> 6, c = idx & 63;
      Bs[c * 40 + k] = (__bf16)Wproj[(size_t)(kk + k) * DIMC + n0 + c];
    }
#if HAVE_ASYNC_LDS
    wait_async0();
#endif
    __syncthreads();
    bf16x16 a = load_a_frag(As, wv * 16 + lc, 40, hf);
    #pragma unroll
    for (int tt = 0; tt < 4; ++tt)
      acc[tt] = wmma_bf16(a, load_bt_frag(Bs, tt * 16 + lc, 40, hf), acc[tt]);
  }
  #pragma unroll
  for (int tt = 0; tt < 4; ++tt) {
    int col = n0 + tt * 16 + lc;
    float bias = bproj[col];
    #pragma unroll
    for (int i = 0; i < 8; ++i) {
      int m = m0 + wv * 16 + i + 8 * hf;
      out[(size_t)m * DIMC + col] = acc[tt][i] + bias;
    }
  }
}

// ---------------------------------------------------------------------------
extern "C" void kernel_launch(void* const* d_in, const int* in_sizes, int n_in,
                              void* d_out, int out_size, void* d_ws, size_t ws_size,
                              hipStream_t stream) {
  (void)in_sizes; (void)n_in; (void)out_size; (void)ws_size;
  const float* x          = (const float*)d_in[0];
  const float* Wqkv       = (const float*)d_in[1];
  const float* Wproj      = (const float*)d_in[2];
  const float* bproj      = (const float*)d_in[3];
  const float* bias_table = (const float*)d_in[4];
  // d_in[5] (rel_index) unused: rel_index[n,m] == m - n + (N-1) analytically.
  float* out = (float*)d_out;

  const size_t plane = (size_t)BB * NH * NPATCH * DH;   // 6,291,456 elems
  __bf16* qkv_ws = (__bf16*)d_ws;                       // q|k|v planes, bf16
  __bf16* o_ws   = qkv_ws + 3 * plane;                  // attention out, bf16

  qkv_gemm_kernel<<<dim3(BB * NPATCH / 64, 3 * DIMC / 64), 256, 0, stream>>>(
      x, Wqkv, qkv_ws);
  flash_attn_kernel<<<dim3(BB * NH * (NPATCH / 64)), 128, 0, stream>>>(
      qkv_ws, bias_table, o_ws);
  proj_gemm_kernel<<<dim3(BB * NPATCH / 64, DIMC / 64), 256, 0, stream>>>(
      o_ws, Wproj, bproj, out);
}
